// DecomposedFrequencyLearning_27135603376578
// MI455X (gfx1250) — matrix operand order repaired and checked by reading, not compiled, not run on hardware
//
#include <hip/hip_runtime.h>

// DecomposedFrequencyLearning on gfx1250:
//   1) NaN forward/backward fill (per-series scan, LDS-transposed I/O)
//   2) truncated real DFT as f16 WMMA GEMM (only f < 416 needed, tau <= 409)
//   3) per-filter masked/weighted spectrum (mu==0 handled as x - lowpass)
//   4) truncated inverse DFT as f16 WMMA GEMM (32x64 tiles)
// GEMM K-loops are software-pipelined with ping-pong fragment buffers
// (manual 2x unroll -> no loop-carried register copies).

typedef _Float16 half8  __attribute__((ext_vector_type(8)));
typedef _Float16 half16 __attribute__((ext_vector_type(16)));
typedef float    float8 __attribute__((ext_vector_type(8)));

#define BDIM   32
#define LDIM   4096
#define CDIM   64
#define SDIM   2048      // B*C series
#define NFREQ  832       // 2 * 416 (cos/sin interleaved), 416 >= 409 max tau
#define KFILT  4

// A fragment (16x32 f16): lane row = l&15, contiguous 16 halfwords at k = 16*half.
__device__ __forceinline__ half16 load_afrag(const _Float16* __restrict__ arow,
                                             int kb, int half) {
  half8 x0 = *(const half8*)(arow + kb + 16 * half);
  half8 x1 = *(const half8*)(arow + kb + 16 * half + 8);
  return __builtin_shufflevector(x0, x1, 0,1,2,3,4,5,6,7,8,9,10,11,12,13,14,15);
}

#define WMMA(A, Bv, Cv) \
  __builtin_amdgcn_wmma_f32_16x16x32_f16(false, (A), false, (Bv), (short)0, (Cv), false, false)

// ---------------------------------------------------------------------------
// Kernel 1: NaN fill. One block per b, 64 threads (one per channel c).
// ---------------------------------------------------------------------------
__global__ __launch_bounds__(64) void k_fill(const float* __restrict__ x,
                                             _Float16* __restrict__ xc) {
  __shared__ float tile[64][65];
  const int b = blockIdx.x;
  const int tid = threadIdx.x;           // 0..63
  float last = 0.0f; bool seen = false;
  // ---- forward pass ----
  for (int t0 = 0; t0 < LDIM; t0 += 64) {
    for (int i = 0; i < 64; ++i)
      tile[i][tid] = x[((size_t)b * LDIM + (t0 + i)) * CDIM + tid];
    __syncthreads();
    for (int i = 0; i < 64; ++i) {
      float v = tile[i][tid];
      if (v == v) { last = v; seen = true; }
      else if (seen) { tile[i][tid] = last; }
    }
    __syncthreads();
    for (int i = 0; i < 64; ++i)
      xc[(size_t)(b * 64 + i) * LDIM + t0 + tid] = (_Float16)tile[tid][i];
    __syncthreads();
  }
  // ---- backward pass (fills leading NaNs; all-NaN series -> 0) ----
  float nxt = 0.0f; bool seenB = false;
  for (int t0 = LDIM - 64; t0 >= 0; t0 -= 64) {
    for (int i = 0; i < 64; ++i)
      tile[tid][i] = (float)xc[(size_t)(b * 64 + i) * LDIM + t0 + tid];
    __syncthreads();
    for (int tl = 63; tl >= 0; --tl) {
      float v = tile[tl][tid];
      if (v == v) { nxt = v; seenB = true; }
      else { tile[tl][tid] = seenB ? nxt : 0.0f; }
    }
    __syncthreads();
    for (int i = 0; i < 64; ++i)
      xc[(size_t)(b * 64 + i) * LDIM + t0 + tid] = (_Float16)tile[tid][i];
    __syncthreads();
  }
}

// ---------------------------------------------------------------------------
// Kernel 2a/2b: DFT basis. basisRow[t][n] (A, inverse GEMM) and basisT[n][t]
// (B, forward GEMM). n=2f -> cos(2*pi*f*t/L), n=2f+1 -> sin.
// ---------------------------------------------------------------------------
__device__ __forceinline__ _Float16 basis_val(int t, int n) {
  int f = n >> 1;
  int ph = (f * t) & (LDIM - 1);
  float th = (float)ph * (6.283185307179586f / (float)LDIM);
  float s, c;
  __sincosf(th, &s, &c);
  return (_Float16)((n & 1) ? s : c);
}

__global__ __launch_bounds__(256) void k_basis_row(_Float16* __restrict__ basisRow) {
  int gid = blockIdx.x * 256 + threadIdx.x;        // over 4096*832
  int n = gid % NFREQ, t = gid / NFREQ;
  basisRow[(size_t)t * NFREQ + n] = basis_val(t, n);
}

__global__ __launch_bounds__(256) void k_basis_t(_Float16* __restrict__ basisT) {
  int gid = blockIdx.x * 256 + threadIdx.x;        // over 832*4096
  int t = gid & (LDIM - 1), n = gid >> 12;
  basisT[(size_t)n * LDIM + t] = basis_val(t, n);
}

// ---------------------------------------------------------------------------
// Kernel 3: forward GEMM  X[s,n] = sum_t xc[s,t] * basisT[n,t]
// One wave per 16(s) x 64(n) tile, K=4096 (128 blocks of 32, even count).
// Ping-pong pipelined: no loop-carried fragment copies.
// ---------------------------------------------------------------------------
__global__ __launch_bounds__(256) void k_fwd(const _Float16* __restrict__ A,
                                             const _Float16* __restrict__ Bt,
                                             float* __restrict__ X) {
  const int wave = blockIdx.x * 8 + (threadIdx.x >> 5);   // 0..1663
  const int lane = threadIdx.x & 31;
  const int st = wave / 13;                               // s tile 0..127
  const int nq = wave % 13;                               // n quad 0..12
  const int m = lane & 15, half = lane >> 4;
  const _Float16* arow = A + (size_t)(st * 16 + m) * LDIM;
  const _Float16* brow = Bt + (size_t)(nq * 64 + m) * LDIM + 16 * half;
  float8 acc[4] = {};

  half16 ap = load_afrag(arow, 0, half);                  // ping
  half16 bp[4];
#pragma unroll
  for (int q = 0; q < 4; ++q)
    bp[q] = *(const half16*)(brow + (size_t)q * 16 * LDIM);
  half16 aq;                                              // pong
  half16 bq[4];

  for (int kb = 0; kb < LDIM - 64; kb += 64) {
    aq = load_afrag(arow, kb + 32, half);
#pragma unroll
    for (int q = 0; q < 4; ++q)
      bq[q] = *(const half16*)(brow + (size_t)q * 16 * LDIM + kb + 32);
#pragma unroll
    for (int q = 0; q < 4; ++q) acc[q] = WMMA(ap, bp[q], acc[q]);

    ap = load_afrag(arow, kb + 64, half);
#pragma unroll
    for (int q = 0; q < 4; ++q)
      bp[q] = *(const half16*)(brow + (size_t)q * 16 * LDIM + kb + 64);
#pragma unroll
    for (int q = 0; q < 4; ++q) acc[q] = WMMA(aq, bq[q], acc[q]);
  }
  // tail: blocks LDIM-64 (in ping) and LDIM-32
  aq = load_afrag(arow, LDIM - 32, half);
#pragma unroll
  for (int q = 0; q < 4; ++q)
    bq[q] = *(const half16*)(brow + (size_t)q * 16 * LDIM + LDIM - 32);
#pragma unroll
  for (int q = 0; q < 4; ++q) acc[q] = WMMA(ap, bp[q], acc[q]);
#pragma unroll
  for (int q = 0; q < 4; ++q) acc[q] = WMMA(aq, bq[q], acc[q]);

  const int srow0 = st * 16 + 8 * half;                   // D rows: m' = r + 8*half
#pragma unroll
  for (int r = 0; r < 8; ++r) {
    size_t o = (size_t)(srow0 + r) * NFREQ + nq * 64 + m;
#pragma unroll
    for (int q = 0; q < 4; ++q) X[o + q * 16] = acc[q][r];
  }
}

// ---------------------------------------------------------------------------
// Kernel 4: masked/weighted spectrum  Yh[k][s][n] = w_k(f) * X[s][n]  (f16)
// w = 1/L (f==0), 2/L (f>=1), 0 (f>=tau_k); negated when mu_k==0 (highpass =
// x_clean - lowpass, added back in kernel 5's epilogue).
// ---------------------------------------------------------------------------
__global__ __launch_bounds__(256) void k_prep(const float* __restrict__ X,
                                              const int* __restrict__ taus,
                                              const int* __restrict__ mus,
                                              _Float16* __restrict__ Yh) {
  int gid = blockIdx.x * 256 + threadIdx.x;        // over 4*2048*832
  int n = gid % NFREQ;
  int s = (gid / NFREQ) % SDIM;
  int k = gid / (NFREQ * SDIM);
  int f = n >> 1;
  float w = (f == 0) ? (1.0f / (float)LDIM) : (2.0f / (float)LDIM);
  if (f >= taus[k]) w = 0.0f;
  if (mus[k] == 0) w = -w;
  Yh[(size_t)gid] = (_Float16)(w * X[(size_t)s * NFREQ + n]);
}

// ---------------------------------------------------------------------------
// Kernel 5: inverse GEMM per (k,b):  out[t,c] = sum_n basisRow[t,n]*Yh[k][b*64+c][n]
// One wave per 32(t) x 64(c) tile (2 A frags share 4 B frags -> 8 WMMA / K-step),
// K=832 (26 blocks of 32, even count), ping-pong pipelined.
// Epilogue adds x_clean when mu==0 (wave-uniform branch).
// ---------------------------------------------------------------------------
__global__ __launch_bounds__(256) void k_inv(const _Float16* __restrict__ basisRow,
                                             const _Float16* __restrict__ Yh,
                                             const _Float16* __restrict__ xc,
                                             const int* __restrict__ mus,
                                             float* __restrict__ out) {
  const int wave = blockIdx.x * 8 + (threadIdx.x >> 5);   // 0..16383
  const int lane = threadIdx.x & 31;
  const int tt = wave & 127;                              // t tile (32 rows)
  const int b  = (wave >> 7) & 31;
  const int k  = wave >> 12;
  const int m = lane & 15, half = lane >> 4;
  const _Float16* arow0 = basisRow + (size_t)(tt * 32 + m) * NFREQ;
  const _Float16* arow1 = basisRow + (size_t)(tt * 32 + 16 + m) * NFREQ;
  const _Float16* yrow  = Yh + (size_t)(k * SDIM + b * 64 + m) * NFREQ + 16 * half;
  float8 acc[8] = {};   // [h*4+q]: h = t half-tile, q = c tile

  half16 a0p = load_afrag(arow0, 0, half);                // ping
  half16 a1p = load_afrag(arow1, 0, half);
  half16 bp[4];
#pragma unroll
  for (int q = 0; q < 4; ++q)
    bp[q] = *(const half16*)(yrow + (size_t)q * 16 * NFREQ);
  half16 a0q, a1q;                                        // pong
  half16 bq[4];

  for (int kb = 0; kb < NFREQ - 64; kb += 64) {
    a0q = load_afrag(arow0, kb + 32, half);
    a1q = load_afrag(arow1, kb + 32, half);
#pragma unroll
    for (int q = 0; q < 4; ++q)
      bq[q] = *(const half16*)(yrow + (size_t)q * 16 * NFREQ + kb + 32);
#pragma unroll
    for (int q = 0; q < 4; ++q) {
      acc[q]     = WMMA(a0p, bp[q], acc[q]);
      acc[4 + q] = WMMA(a1p, bp[q], acc[4 + q]);
    }
    a0p = load_afrag(arow0, kb + 64, half);
    a1p = load_afrag(arow1, kb + 64, half);
#pragma unroll
    for (int q = 0; q < 4; ++q)
      bp[q] = *(const half16*)(yrow + (size_t)q * 16 * NFREQ + kb + 64);
#pragma unroll
    for (int q = 0; q < 4; ++q) {
      acc[q]     = WMMA(a0q, bq[q], acc[q]);
      acc[4 + q] = WMMA(a1q, bq[q], acc[4 + q]);
    }
  }
  // tail: blocks NFREQ-64 (in ping) and NFREQ-32
  a0q = load_afrag(arow0, NFREQ - 32, half);
  a1q = load_afrag(arow1, NFREQ - 32, half);
#pragma unroll
  for (int q = 0; q < 4; ++q)
    bq[q] = *(const half16*)(yrow + (size_t)q * 16 * NFREQ + NFREQ - 32);
#pragma unroll
  for (int q = 0; q < 4; ++q) {
    acc[q]     = WMMA(a0p, bp[q], acc[q]);
    acc[4 + q] = WMMA(a1p, bp[q], acc[4 + q]);
  }
#pragma unroll
  for (int q = 0; q < 4; ++q) {
    acc[q]     = WMMA(a0q, bq[q], acc[q]);
    acc[4 + q] = WMMA(a1q, bq[q], acc[4 + q]);
  }

  const int mu = mus[k];                 // wave-uniform
#pragma unroll
  for (int h = 0; h < 2; ++h) {
    const int t0 = tt * 32 + h * 16 + 8 * half;   // D rows: t = t0 + r
#pragma unroll
    for (int q = 0; q < 4; ++q) {
      const int c = q * 16 + m;
      float addv[8];
#pragma unroll
      for (int r = 0; r < 8; ++r) addv[r] = 0.0f;
      if (mu == 0) {
        half8 xv = *(const half8*)(xc + (size_t)(b * 64 + c) * LDIM + t0);
#pragma unroll
        for (int r = 0; r < 8; ++r) addv[r] = (float)xv[r];
      }
      size_t obase = ((size_t)(k * BDIM + b) * LDIM + t0) * CDIM + c;
#pragma unroll
      for (int r = 0; r < 8; ++r)
        out[obase + (size_t)r * CDIM] = acc[h * 4 + q][r] + addv[r];
    }
  }
}

// ---------------------------------------------------------------------------
extern "C" void kernel_launch(void* const* d_in, const int* in_sizes, int n_in,
                              void* d_out, int out_size, void* d_ws, size_t ws_size,
                              hipStream_t stream) {
  (void)in_sizes; (void)n_in; (void)out_size; (void)ws_size;
  const float* x    = (const float*)d_in[0];
  const int*   taus = (const int*)d_in[1];
  const int*   mus  = (const int*)d_in[2];
  float*       out  = (float*)d_out;

  // workspace carve-out (256B aligned); total ~50.9 MB
  char* ws = (char*)d_ws;
  size_t off = 0;
  auto carve = [&](size_t bytes) { char* p = ws + off; off = (off + bytes + 255) & ~(size_t)255; return p; };
  _Float16* xc       = (_Float16*)carve((size_t)SDIM * LDIM * 2);          // 16 MB
  _Float16* basisRow = (_Float16*)carve((size_t)LDIM * NFREQ * 2);         // 6.8 MB
  _Float16* basisT   = (_Float16*)carve((size_t)NFREQ * LDIM * 2);         // 6.8 MB
  float*    Xf       = (float*)   carve((size_t)SDIM * NFREQ * 4);         // 6.8 MB
  _Float16* Yh       = (_Float16*)carve((size_t)KFILT * SDIM * NFREQ * 2); // 13.6 MB

  k_fill<<<BDIM, 64, 0, stream>>>(x, xc);
  k_basis_row<<<(LDIM * NFREQ) / 256, 256, 0, stream>>>(basisRow);
  k_basis_t<<<(NFREQ * LDIM) / 256, 256, 0, stream>>>(basisT);
  k_fwd<<<(128 * 13) / 8, 256, 0, stream>>>(xc, basisT, Xf);               // 1664 waves
  k_prep<<<(KFILT * SDIM * NFREQ) / 256, 256, 0, stream>>>(Xf, taus, mus, Yh);
  k_inv<<<(KFILT * BDIM * 128) / 8, 256, 0, stream>>>(basisRow, Yh, xc, mus, out); // 16384 waves
}
